// AdditiveMetaAttentionBlock_30726196035632
// MI455X (gfx1250) — matrix operand order repaired and checked by reading, not compiled, not run on hardware
//
#include <hip/hip_runtime.h>
#include <hip/hip_bf16.h>

#define N_  32
#define C_  256
#define H_  64
#define W_  64
#define HW_ 4096
#define E_  256
#define DM_ 128
#define DG_ 512
#define EPS_ 1e-5f

typedef __attribute__((ext_vector_type(16))) __bf16 v16bf;
typedef __attribute__((ext_vector_type(8)))  float  v8f;
typedef __attribute__((ext_vector_type(4)))  float  v4f;
typedef __attribute__((ext_vector_type(2)))  float  v2f;

union Frag16 { v16bf v; unsigned u[8]; };

#if __has_builtin(__builtin_amdgcn_tanhf)
#define TANHF(x) __builtin_amdgcn_tanhf(x)   // native V_TANH_F32 on gfx1250
#else
#define TANHF(x) tanhf(x)
#endif

__device__ __forceinline__ unsigned short f2bf(float f) {
    union { float f; unsigned u; } v; v.f = f;
    unsigned r = v.u + 0x7FFFu + ((v.u >> 16) & 1u);   // round-to-nearest-even
    return (unsigned short)(r >> 16);
}

// ---------------------------------------------------------------- prep: Wpw f32 -> bf16
__global__ void k_cvt_wpw(const float* __restrict__ Wpw, unsigned short* __restrict__ Wb) {
    int i = blockIdx.x * blockDim.x + threadIdx.x;
    if (i < E_ * C_) Wb[i] = f2bf(Wpw[i]);
}

// ---------------------------------------------------------------- bias(n,e) = meta_embed + g_embed
__global__ void k_bias(const float* __restrict__ Xm, const float* __restrict__ G,
                       const float* __restrict__ Wm, const float* __restrict__ Wg,
                       float* __restrict__ bias) {
    int n = blockIdx.x, e = threadIdx.x;
    const float* xm = Xm + n * DM_;
    const float* g  = G  + n * DG_;
    const float* wm = Wm + e * DM_;
    const float* wg = Wg + e * DG_;
    float s = 0.f;
    for (int d = 0; d < DM_; ++d) s += xm[d] * wm[d];
    for (int d = 0; d < DG_; ++d) s += g[d]  * wg[d];
    bias[n * E_ + e] = s;
}

// ---------------------------------------------------------------- fused dwconv + WMMA pointwise + tanh-score
// block = (n, h); 256 threads = 8 waves
__global__ void __launch_bounds__(256)
k_main(const float* __restrict__ X, const float* __restrict__ Wdw,
       const unsigned* __restrict__ Wb32,   // Wpw in bf16, 2 per u32, row-major (E,C)
       const float* __restrict__ bias, const float* __restrict__ Ws,
       float* __restrict__ score) {
    __shared__ unsigned lds_dw[128 * 64];   // 32KB: (c-pair, w) -> {bf16 dw[2t][w], bf16 dw[2t+1][w]}
    __shared__ float    lds_score[64];

    const int tid = threadIdx.x;
    const int n = blockIdx.x >> 6;
    const int h = blockIdx.x & 63;

    if (tid < 64) lds_score[tid] = 0.f;

    // ---- phase 1: depthwise 3x3 conv; each thread makes a PAIR of adjacent w outputs
    //      (v2f arithmetic -> packed/dual-issue fp32 FMA)
    for (int idx = tid; idx < C_ * (W_ / 2); idx += 256) {
        int c = idx >> 5, w = (idx & 31) * 2;
        const float* wk = Wdw + c * 9;
        const float* base = X + (size_t)((n * C_ + c) * H_) * W_;
        v2f acc = {0.f, 0.f};
#pragma unroll
        for (int dy = -1; dy <= 1; ++dy) {
            int hh = h + dy;
            if (hh < 0 || hh >= H_) continue;
            const float* row = base + hh * W_;
#pragma unroll
            for (int dx = -1; dx <= 1; ++dx) {
                float k = wk[(dy + 1) * 3 + (dx + 1)];
                int w0 = w + dx;                       // w0 in [-1, 63]
                float ax = (w0 >= 0)      ? row[w0]     : 0.f;
                float ay = (w0 + 1 <= 63) ? row[w0 + 1] : 0.f;
                v2f val = {ax, ay};
                acc += val * k;
            }
        }
        unsigned short* sh = reinterpret_cast<unsigned short*>(lds_dw);
        int bi = ((c >> 1) * 64 + w) * 2 + (c & 1);
        sh[bi]     = f2bf(acc.x);
        sh[bi + 2] = f2bf(acc.y);
    }
    __syncthreads();

    // ---- phase 2: Y(E=256, W=64) = Wpw @ dw, bf16 WMMA with f32 accumulate
    const int wave = tid >> 5, lane = tid & 31;
    const int lo = lane & 15, hi = lane >> 4;

    v8f acc[2][4] = {};
#pragma unroll
    for (int kk = 0; kk < 8; ++kk) {              // K = 256, 32 per step
        Frag16 a[2];
#pragma unroll
        for (int mi = 0; mi < 2; ++mi) {
            int e = (wave * 2 + mi) * 16 + lo;    // A: lane 0-15 -> M rows
#pragma unroll
            for (int i = 0; i < 8; ++i) {
                int k = kk * 32 + (i >> 2) * 16 + hi * 8 + (i & 3) * 2;
                a[mi].u[i] = Wb32[e * (C_ / 2) + (k >> 1)];
            }
        }
        Frag16 b[4];
#pragma unroll
        for (int ni = 0; ni < 4; ++ni) {
            int w = ni * 16 + lo;                 // B: lane 0-15 -> N cols
#pragma unroll
            for (int j = 0; j < 8; ++j) {
                int t = kk * 16 + (j >> 2) * 8 + hi * 4 + (j & 3);     // c-pair index
                b[ni].u[j] = lds_dw[t * 64 + w];
            }
        }
#pragma unroll
        for (int mi = 0; mi < 2; ++mi)
#pragma unroll
            for (int ni = 0; ni < 4; ++ni)
                acc[mi][ni] = __builtin_amdgcn_wmma_f32_16x16x32_bf16(
                    false, a[mi].v, false, b[ni].v, (short)0, acc[mi][ni], false, false);
    }

    // ---- phase 3: score(p) = sum_e Ws[e] * tanh(Y[e,p] + bias[n,e])
    const float* bs = bias + n * E_;
#pragma unroll
    for (int ni = 0; ni < 4; ++ni) {
        float s = 0.f;
#pragma unroll
        for (int mi = 0; mi < 2; ++mi) {
#pragma unroll
            for (int r = 0; r < 8; ++r) {
                int e = (wave * 2 + mi) * 16 + hi * 8 + r;   // C/D layout: M = r + 8*(lane/16)
                float v = acc[mi][ni][r] + bs[e];
                s += TANHF(v) * Ws[e];
            }
        }
        atomicAdd(&lds_score[ni * 16 + lo], s);
    }
    __syncthreads();
    if (tid < 64) score[n * HW_ + h * 64 + tid] = lds_score[tid];
}

// ---------------------------------------------------------------- softmax over hw per n
__global__ void k_softmax(const float* __restrict__ score, float* __restrict__ attn) {
    __shared__ float red[256];
    int n = blockIdx.x, tid = threadIdx.x;
    const float* sc = score + n * HW_;
    float* at = attn + n * HW_;

    float m = -1e30f;
    for (int i = tid; i < HW_; i += 256) m = fmaxf(m, sc[i]);
    red[tid] = m; __syncthreads();
    for (int s = 128; s > 0; s >>= 1) { if (tid < s) red[tid] = fmaxf(red[tid], red[tid + s]); __syncthreads(); }
    float M = red[0]; __syncthreads();

    float sum = 0.f;
    for (int i = tid; i < HW_; i += 256) { float e = __expf(sc[i] - M); at[i] = e; sum += e; }
    red[tid] = sum; __syncthreads();
    for (int s = 128; s > 0; s >>= 1) { if (tid < s) red[tid] += red[tid + s]; __syncthreads(); }
    float inv = 1.f / red[0];
    for (int i = tid; i < HW_; i += 256) at[i] *= inv;
}

// ---------------------------------------------------------------- context(n,c) = sum_p attn(n,p) * X(n,c,p)
// block = (n, channel-group of 32); wave-per-channel, float4 (b128) coalesced reads
__global__ void k_ctx(const float* __restrict__ X, const float* __restrict__ attn,
                      float* __restrict__ ctx) {
    __shared__ float red[256];
    int n = blockIdx.x >> 3, cg = blockIdx.x & 7;
    int tid = threadIdx.x, wave = tid >> 5, lane = tid & 31;
    const v4f* at4 = (const v4f*)(attn + n * HW_);
#pragma unroll
    for (int cl = 0; cl < 4; ++cl) {
        int c = cg * 32 + wave * 4 + cl;
        const v4f* xr4 = (const v4f*)(X + (size_t)(n * C_ + c) * HW_);
        float p = 0.f;
        for (int i = lane; i < HW_ / 4; i += 32) {
            v4f a = at4[i], x = xr4[i];
            p += a.x * x.x + a.y * x.y + a.z * x.z + a.w * x.w;
        }
        red[tid] = p;
        __syncthreads();
        if (lane == 0) {
            float s = 0.f;
            for (int l = 0; l < 32; ++l) s += red[wave * 32 + l];
            ctx[n * C_ + c] = s;
        }
        __syncthreads();
    }
}

// ---------------------------------------------------------------- LayerNorm over C per n
__global__ void k_ln(const float* __restrict__ ctx, const float* __restrict__ gamma,
                     const float* __restrict__ beta, float* __restrict__ out) {
    __shared__ float red[256];
    int n = blockIdx.x, tid = threadIdx.x;
    float v = ctx[n * C_ + tid];
    red[tid] = v; __syncthreads();
    for (int s = 128; s > 0; s >>= 1) { if (tid < s) red[tid] += red[tid + s]; __syncthreads(); }
    float mu = red[0] * (1.f / C_); __syncthreads();
    float d = v - mu;
    red[tid] = d * d; __syncthreads();
    for (int s = 128; s > 0; s >>= 1) { if (tid < s) red[tid] += red[tid + s]; __syncthreads(); }
    float var = red[0] * (1.f / C_);
    out[n * C_ + tid] = d * rsqrtf(var + EPS_) * gamma[tid] + beta[tid];
}

extern "C" void kernel_launch(void* const* d_in, const int* in_sizes, int n_in,
                              void* d_out, int out_size, void* d_ws, size_t ws_size,
                              hipStream_t stream) {
    const float* X_img = (const float*)d_in[0];
    const float* X_meta = (const float*)d_in[1];
    const float* G      = (const float*)d_in[2];
    const float* Wm     = (const float*)d_in[3];
    const float* Wdw    = (const float*)d_in[4];
    const float* Wpw    = (const float*)d_in[5];
    const float* Wg     = (const float*)d_in[6];
    const float* Ws     = (const float*)d_in[7];
    const float* gamma  = (const float*)d_in[8];
    const float* beta   = (const float*)d_in[9];

    // d_out: context (N*C floats) then attn (N*HW floats)
    float* out_ctx  = (float*)d_out;
    float* out_attn = (float*)d_out + N_ * C_;

    // scratch layout
    char* ws = (char*)d_ws;
    unsigned short* Wb = (unsigned short*)ws;            ws += E_ * C_ * sizeof(unsigned short);  // 128KB
    float* bias  = (float*)ws;                           ws += N_ * E_ * sizeof(float);           // 32KB
    float* score = (float*)ws;                           ws += N_ * HW_ * sizeof(float);          // 512KB
    float* ctx   = (float*)ws;                           ws += N_ * C_ * sizeof(float);           // 32KB
    (void)ws_size; (void)in_sizes; (void)n_in; (void)out_size;

    k_cvt_wpw<<<(E_ * C_ + 255) / 256, 256, 0, stream>>>(Wpw, Wb);
    k_bias<<<N_, E_, 0, stream>>>(X_meta, G, Wm, Wg, bias);
    k_main<<<N_ * H_, 256, 0, stream>>>(X_img, Wdw, (const unsigned*)Wb, bias, Ws, score);
    k_softmax<<<N_, 256, 0, stream>>>(score, out_attn);
    k_ctx<<<N_ * 8, 256, 0, stream>>>(X_img, out_attn, ctx);
    k_ln<<<N_, 256, 0, stream>>>(ctx, gamma, beta, out_ctx);
}
